// GAT_6640019439793
// MI455X (gfx1250) — matrix operand order repaired and checked by reading, not compiled, not run on hardware
//
#include <hip/hip_runtime.h>
#include <hip/hip_bf16.h>

typedef __attribute__((ext_vector_type(16))) _Float16 v16h;
typedef __attribute__((ext_vector_type(8)))  float    v8f;

#define F0   128
#define HDIM 64     // HEADS*HID
#define NHEAD 4
#define HID  16
#define NCLS 10

// ---------------------------------------------------------------------------
// Fused WMMA GEMM + attention dots:
//   H[M x 64] = X[M x K] * W[K x 64]            (K = 128 or 64, M % 16 == 0)
//   aS[n,h]   = <H[n, h*16 : h*16+16], attS[h]>  (same for aD)
// Block = 128 threads = 4 waves; wave w computes the 16-col tile w (== head w)
// for row tile blockIdx.x.  W is staged TRANSPOSED in LDS as f16 so each
// lane's B fragment is 16 contiguous halves (2x ds_load_b128).  K templated
// so the chunk loop fully unrolls -> back-to-back v_wmma issues.
// C/D VGPR layout per CDNA5 ISA 7.12.2; A/B use contiguous k = 16*half + s, not the ISA table.
// ---------------------------------------------------------------------------
template <int K>
__global__ void __launch_bounds__(128)
k_gemm_att(const float* __restrict__ X, const float* __restrict__ W,
           const float* __restrict__ attS, const float* __restrict__ attD,
           float* __restrict__ H, float* __restrict__ aS, float* __restrict__ aD) {
    __shared__ __attribute__((aligned(32))) _Float16 sWT[HDIM * K]; // [col][k]
    const int tid = threadIdx.x;
    for (int i = tid; i < K * HDIM; i += 128) {          // contiguous LDS writes
        int c = i / K, k = i % K;
        sWT[i] = (_Float16)W[k * HDIM + c];
    }
    __syncthreads();

    const int wave  = tid >> 5;             // 0..3 -> column tile == head
    const int lane  = tid & 31;
    const int l16   = lane & 15;
    const int half  = lane >> 4;
    const int mbase = blockIdx.x * 16;
    const int col   = wave * 16 + l16;      // B/C/D column
    const int arow  = mbase + l16;          // A row

    v8f acc = {};
    #pragma unroll
    for (int kc = 0; kc < K; kc += 32) {
        // A-matrix 16x32 f16 layout: lane-half h holds K = kc+16h..kc+16h+15
        // in elements 0..15 (matches B) -> one contiguous 64B float group.
        const float* xr = X + (size_t)arow * K + kc;
        const float4 x0 = ((const float4*)(xr + 16 * half))[0];
        const float4 x1 = ((const float4*)(xr + 16 * half))[1];
        const float4 x2 = ((const float4*)(xr + 16 * half + 8))[0];
        const float4 x3 = ((const float4*)(xr + 16 * half + 8))[1];
        v16h a;
        a[0]  = (_Float16)x0.x; a[1]  = (_Float16)x0.y;
        a[2]  = (_Float16)x0.z; a[3]  = (_Float16)x0.w;
        a[4]  = (_Float16)x1.x; a[5]  = (_Float16)x1.y;
        a[6]  = (_Float16)x1.z; a[7]  = (_Float16)x1.w;
        a[8]  = (_Float16)x2.x; a[9]  = (_Float16)x2.y;
        a[10] = (_Float16)x2.z; a[11] = (_Float16)x2.w;
        a[12] = (_Float16)x3.x; a[13] = (_Float16)x3.y;
        a[14] = (_Float16)x3.z; a[15] = (_Float16)x3.w;
        // B-matrix 32x16: lane-half h holds K = kc+16h..kc+16h+15 for column
        // `col` -> 16 contiguous halves in transposed LDS image.
        const v16h b = *(const v16h*)(sWT + (size_t)col * K + kc + 16 * half);

        acc = __builtin_amdgcn_wmma_f32_16x16x32_f16(
            /*neg_a=*/false, a, /*neg_b=*/false, b,
            /*c_mod=*/(short)0, acc, /*reuse_a=*/false, /*reuse_b=*/false);
    }

    // Epilogue: store H tile; fused per-head attention dots via 16-lane
    // xor-shuffle reduction (C/D layout: lane holds (M = r + 8*half, N = l16)).
    const float as_l = attS[wave * HID + l16];
    const float ad_l = attD[wave * HID + l16];
    #pragma unroll
    for (int r = 0; r < 8; ++r) {
        const int row = mbase + r + 8 * half;
        H[(size_t)row * HDIM + col] = acc[r];
        float vs = acc[r] * as_l;
        float vd = acc[r] * ad_l;
        #pragma unroll
        for (int m = 8; m >= 1; m >>= 1) {
            vs += __shfl_xor(vs, m, 32);
            vd += __shfl_xor(vd, m, 32);
        }
        if (l16 == 0) {
            aS[row * NHEAD + wave] = vs;
            aD[row * NHEAD + wave] = vd;
        }
    }
}

__device__ __forceinline__ void edge_sd(const int* __restrict__ ei, int e, int E,
                                        int& s, int& d) {
    if (e < E) { s = ei[e]; d = ei[E + e]; } else { s = e - E; d = e - E; }
}

// sign-aware float atomic max (buffer initialized to -inf bit pattern)
__device__ __forceinline__ void atomicMaxF(float* addr, float val) {
    if (val >= 0.f) atomicMax((int*)addr, __float_as_int(val));
    else            atomicMin((unsigned int*)addr, __float_as_uint(val));
}

// pass 1: leaky-relu logit -> ev, segment max -> m
__global__ void k_edge_logit(const int* __restrict__ ei, int E, int ET,
                             const float* __restrict__ aS, const float* __restrict__ aD,
                             float* __restrict__ ev, float* __restrict__ m) {
    int total = ET * 4;
    for (int i = blockIdx.x * blockDim.x + threadIdx.x; i < total;
         i += gridDim.x * blockDim.x) {
        int e = i >> 2, h = i & 3, s, d;
        edge_sd(ei, e, E, s, d);
        float l = aS[s * 4 + h] + aD[d * 4 + h];
        l = l > 0.f ? l : 0.2f * l;
        ev[i] = l;
        atomicMaxF(&m[d * 4 + h], l);
    }
}

// pass 2: ev = exp(logit - m[dst]); segment sum -> ssum
__global__ void k_edge_expsum(const int* __restrict__ ei, int E, int ET,
                              const float* __restrict__ m, float* __restrict__ ev,
                              float* __restrict__ ssum) {
    int total = ET * 4;
    for (int i = blockIdx.x * blockDim.x + threadIdx.x; i < total;
         i += gridDim.x * blockDim.x) {
        int e = i >> 2, h = i & 3, s, d;
        edge_sd(ei, e, E, s, d);
        float x = __expf(ev[i] - m[d * 4 + h]);
        ev[i] = x;
        atomicAdd(&ssum[d * 4 + h], x);
    }
}

// pass 3: out[dst,:] += h[src,:] * alpha
__global__ void k_edge_agg(const int* __restrict__ ei, int E, int ET,
                           const float* __restrict__ H, const float* __restrict__ ev,
                           const float* __restrict__ ssum, float* __restrict__ out) {
    int total = ET * HDIM;
    for (int i = blockIdx.x * blockDim.x + threadIdx.x; i < total;
         i += gridDim.x * blockDim.x) {
        int e = i >> 6, c = i & 63, h = c >> 4, s, d;
        edge_sd(ei, e, E, s, d);
        float alpha = ev[e * 4 + h] / (ssum[d * 4 + h] + 1e-16f);
        atomicAdd(&out[(size_t)d * HDIM + c], H[(size_t)s * HDIM + c] * alpha);
    }
}

__global__ void k_bias_silu(float* __restrict__ h, const float* __restrict__ b,
                            int total /* N*64 */) {
    for (int i = blockIdx.x * blockDim.x + threadIdx.x; i < total;
         i += gridDim.x * blockDim.x) {
        float v = h[i] + b[i & 63];
        h[i] = v / (1.f + __expf(-v));
    }
}

__global__ void k_pool(const float* __restrict__ h, const int* __restrict__ batch,
                       float* __restrict__ pooled, int total /* N*64 */) {
    for (int i = blockIdx.x * blockDim.x + threadIdx.x; i < total;
         i += gridDim.x * blockDim.x) {
        int n = i >> 6, c = i & 63;
        atomicAdd(&pooled[(size_t)batch[n] * HDIM + c], h[i]);
    }
}

__global__ void k_head(const float* __restrict__ pooled, const float* __restrict__ Wr,
                       const float* __restrict__ br, float* __restrict__ out, int NG) {
    int g = blockIdx.x * blockDim.x + threadIdx.x;
    if (g >= NG) return;
    float z[NCLS];
    float mx = -1e30f;
    #pragma unroll
    for (int c = 0; c < NCLS; ++c) {
        float acc = br[c];
        for (int k = 0; k < HDIM; ++k) acc += pooled[g * HDIM + k] * Wr[k * NCLS + c];
        acc = acc > 0.f ? acc : 0.f;           // make_mlp ends with ReLU
        z[c] = acc;
        mx = fmaxf(mx, acc);
    }
    float se = 0.f;
    #pragma unroll
    for (int c = 0; c < NCLS; ++c) se += __expf(z[c] - mx);
    float ls = __logf(se);
    #pragma unroll
    for (int c = 0; c < NCLS; ++c) out[g * NCLS + c] = z[c] - mx - ls;
}

__global__ void k_fillf(float* __restrict__ p, int n, float v) {
    for (int i = blockIdx.x * blockDim.x + threadIdx.x; i < n;
         i += gridDim.x * blockDim.x)
        p[i] = v;
}

// ---------------------------------------------------------------------------
extern "C" void kernel_launch(void* const* d_in, const int* in_sizes, int n_in,
                              void* d_out, int out_size, void* d_ws, size_t ws_size,
                              hipStream_t stream) {
    const float* x     = (const float*)d_in[0];
    const int*   ei    = (const int*)d_in[1];
    const int*   batch = (const int*)d_in[2];
    const float* W[3]  = {(const float*)d_in[3],  (const float*)d_in[7],  (const float*)d_in[11]};
    const float* B[3]  = {(const float*)d_in[4],  (const float*)d_in[8],  (const float*)d_in[12]};
    const float* AS[3] = {(const float*)d_in[5],  (const float*)d_in[9],  (const float*)d_in[13]};
    const float* AD[3] = {(const float*)d_in[6],  (const float*)d_in[10], (const float*)d_in[14]};
    const float* Wr    = (const float*)d_in[15];
    const float* br    = (const float*)d_in[16];

    const int N  = in_sizes[0] / F0;      // 50000 (multiple of 16)
    const int E  = in_sizes[1] / 2;       // 800000
    const int ET = E + N;                 // + self loops
    const int NG = out_size / NCLS;       // 128

    // carve workspace (all 256B aligned)
    char* p = (char*)d_ws;
    auto carve = [&](size_t bytes) -> float* {
        float* r = (float*)p;
        p += (bytes + 255) & ~(size_t)255;
        return r;
    };
    float* hA     = carve((size_t)N * HDIM * 4);   // GEMM output (layer features)
    float* hB     = carve((size_t)N * HDIM * 4);   // aggregation output
    float* aS     = carve((size_t)N * 4 * 4);
    float* aD     = carve((size_t)N * 4 * 4);
    float* mMax   = carve((size_t)N * 4 * 4);
    float* sSum   = carve((size_t)N * 4 * 4);
    float* ev     = carve((size_t)ET * 4 * 4);
    float* pooled = carve((size_t)NG * HDIM * 4);

    const int TB = 256;
    const int gNH  = (N * 4 + TB - 1) / TB;
    const int gE4  = (ET * 4 + TB - 1) / TB;
    const int gE64 = (ET * HDIM + TB - 1) / TB;
    const int gN64 = (N * HDIM + TB - 1) / TB;
    const float ninf = -__builtin_inff();

    const float* cur = x;
    for (int l = 0; l < 3; ++l) {
        if (l == 0)
            k_gemm_att<F0>  <<<N / 16, 128, 0, stream>>>(cur, W[l], AS[l], AD[l], hA, aS, aD);
        else
            k_gemm_att<HDIM><<<N / 16, 128, 0, stream>>>(cur, W[l], AS[l], AD[l], hA, aS, aD);
        k_fillf      <<<gNH, TB, 0, stream>>>(mMax, N * 4, ninf);
        k_fillf      <<<gNH, TB, 0, stream>>>(sSum, N * 4, 0.f);
        k_edge_logit <<<gE4, TB, 0, stream>>>(ei, E, ET, aS, aD, ev, mMax);
        k_edge_expsum<<<gE4, TB, 0, stream>>>(ei, E, ET, mMax, ev, sSum);
        k_fillf      <<<gN64, TB, 0, stream>>>(hB, N * HDIM, 0.f);
        k_edge_agg   <<<gE64, TB, 0, stream>>>(ei, E, ET, hA, ev, sSum, hB);
        k_bias_silu  <<<gN64, TB, 0, stream>>>(hB, B[l], N * HDIM);
        cur = hB;
    }
    k_fillf<<<(NG * HDIM + TB - 1) / TB, TB, 0, stream>>>(pooled, NG * HDIM, 0.f);
    k_pool <<<gN64, TB, 0, stream>>>(hB, batch, pooled, N * HDIM);
    k_head <<<(NG + 127) / 128, 128, 0, stream>>>(pooled, Wr, br, (float*)d_out, NG);
}
